// Backbone_26121991094961
// MI455X (gfx1250) — matrix operand order repaired and checked by reading, not compiled, not run on hardware
//
#include <hip/hip_runtime.h>
#include <hip/hip_bf16.h>

// ---------------------------------------------------------------------------
// Point-Transformer backbone for MI455X (gfx1250, wave32, WMMA).
// All dense linear algebra runs on v_wmma_f32_16x16x32_bf16 (bf16 in, f32 acc).
// ---------------------------------------------------------------------------

typedef __bf16 bf16_t;
typedef __attribute__((ext_vector_type(16))) __bf16 v16bf;
typedef __attribute__((ext_vector_type(8)))  __bf16 v8bf;
typedef __attribute__((ext_vector_type(4)))  __bf16 v4bf;
typedef __attribute__((ext_vector_type(8)))  float  v8f;
typedef __attribute__((ext_vector_type(4)))  float  v4f;

__device__ __forceinline__ v8f zero8() {
  v8f z;
#pragma unroll
  for (int i = 0; i < 8; ++i) z[i] = 0.f;
  return z;
}

__device__ __forceinline__ v16bf mk_frag(v8bf lo, v8bf hi) {
  v16bf r;
#pragma unroll
  for (int e = 0; e < 8; ++e) { r[e] = lo[e]; r[8 + e] = hi[e]; }
  return r;
}

// ---------------------------------------------------------------------------
// WMMA GEMM:  C = epi(alpha * A*B + bias [, residual])
//   Template-specialized on dtypes/layout so staging has no runtime branches.
//   Block tile 128x64, K-step 32, 8 waves, each wave owns 32x32 (2x2 WMMA).
//   AL=true fast path: unguarded vector loads, register-staged pipeline
//   (next tile's global loads issued before the WMMAs of the current tile),
//   guard-free epilogue.
//   epi: 0 = none, 1 = relu(v), 2 = relu(residual + relu(v))
// ---------------------------------------------------------------------------
#define GEMM_TM 128
#define GEMM_TN 64
#define GEMM_TK 32
#define GEMM_LK (GEMM_TK + 8)   // padded LDS row: 80B stride -> conflict-free frags

template<bool ABF, bool BBF, bool TRB, bool AL>
__global__ __launch_bounds__(256) void gemm_wmma_t(
    const void* __restrict__ Aptr, int lda,
    const void* __restrict__ Bptr, int ldb,
    const float* __restrict__ bias,
    const float* __restrict__ residual, int ldres,
    void* __restrict__ Cptr, int c_bf16, int ldc,
    int M, int N, int Kdim, float alpha, int epi)
{
  __shared__ bf16_t lA[GEMM_TM][GEMM_LK];   // 10 KB
  __shared__ bf16_t lB[GEMM_TN][GEMM_LK];   // 5 KB   (stored [n][k])

  const int tid  = threadIdx.x;
  const int wv   = tid >> 5;
  const int lane = tid & 31;
  const int m16  = lane & 15;
  const int hi8  = (lane >> 4) * 8;     // A-frag K sub-base / C-row sub-base
  const int kB   = (lane >> 4) * 16;    // unused here; B-frag uses hi8 like A
  const int wr   = wv >> 1;             // 0..3
  const int wc   = wv & 1;              // 0..1
  const int bM   = blockIdx.y * GEMM_TM;
  const int bN   = blockIdx.x * GEMM_TN;

  v8f acc[2][2];
#pragma unroll
  for (int i = 0; i < 2; ++i)
#pragma unroll
    for (int j = 0; j < 2; ++j) acc[i][j] = zero8();

  if constexpr (AL) {
    // ---------------- aligned, register-staged pipeline ----------------
    v8bf   raB[2];  v4f   raF[4];                   // A staging regs
    v8bf   rbB;     bf16_t rbS[8];  float rbF[8];   // B staging regs

    auto loadA = [&](int k0) {
      if constexpr (ABF) {
#pragma unroll
        for (int u = 0; u < 2; ++u) {
          int c = tid + u * 256; int r = c >> 2, kv = (c & 3) * 8;
          raB[u] = *(const v8bf*)((const bf16_t*)Aptr + (size_t)(bM + r) * lda + k0 + kv);
        }
      } else {
#pragma unroll
        for (int u = 0; u < 4; ++u) {
          int c = tid + u * 256; int r = c >> 3, kv = (c & 7) * 4;
          raF[u] = *(const v4f*)((const float*)Aptr + (size_t)(bM + r) * lda + k0 + kv);
        }
      }
    };
    auto storeA = [&]() {
      if constexpr (ABF) {
#pragma unroll
        for (int u = 0; u < 2; ++u) {
          int c = tid + u * 256; int r = c >> 2, kv = (c & 3) * 8;
          *(v8bf*)&lA[r][kv] = raB[u];
        }
      } else {
#pragma unroll
        for (int u = 0; u < 4; ++u) {
          int c = tid + u * 256; int r = c >> 3, kv = (c & 7) * 4;
          v4bf t;
#pragma unroll
          for (int j = 0; j < 4; ++j) t[j] = (bf16_t)raF[u][j];
          *(v4bf*)&lA[r][kv] = t;
        }
      }
    };
    auto loadB = [&](int k0) {
      if constexpr (TRB && BBF) {         // e.g. S = Q K^T: row n contiguous in k
        int n = tid >> 2, kv = (tid & 3) * 8;
        rbB = *(const v8bf*)((const bf16_t*)Bptr + (size_t)(bN + n) * ldb + k0 + kv);
      } else if constexpr (BBF) {         // [k][n] bf16: scalar, coalesced along n
#pragma unroll
        for (int r = 0; r < 8; ++r) {
          int e = tid + r * 256; int k = e >> 6, n = e & 63;
          rbS[r] = ((const bf16_t*)Bptr)[(size_t)(k0 + k) * ldb + bN + n];
        }
      } else {                            // [k][n] f32 weights: coalesced along n
#pragma unroll
        for (int r = 0; r < 8; ++r) {
          int e = tid + r * 256; int k = e >> 6, n = e & 63;
          size_t idx = TRB ? ((size_t)(bN + n) * ldb + k0 + k)
                           : ((size_t)(k0 + k) * ldb + bN + n);
          rbF[r] = ((const float*)Bptr)[idx];
        }
      }
    };
    auto storeB = [&]() {
      if constexpr (TRB && BBF) {
        int n = tid >> 2, kv = (tid & 3) * 8;
        *(v8bf*)&lB[n][kv] = rbB;
      } else if constexpr (BBF) {
#pragma unroll
        for (int r = 0; r < 8; ++r) {
          int e = tid + r * 256; int k = e >> 6, n = e & 63;
          lB[n][k] = rbS[r];
        }
      } else {
#pragma unroll
        for (int r = 0; r < 8; ++r) {
          int e = tid + r * 256; int k = e >> 6, n = e & 63;
          lB[n][k] = (bf16_t)rbF[r];
        }
      }
    };

    loadA(0); loadB(0);
    for (int k0 = 0; k0 < Kdim; k0 += GEMM_TK) {
      storeA(); storeB();
      __syncthreads();
      if (k0 + GEMM_TK < Kdim) { loadA(k0 + GEMM_TK); loadB(k0 + GEMM_TK); }

      v16bf af[2], bfr[2];
#pragma unroll
      for (int t = 0; t < 2; ++t) {
        const bf16_t* pa = &lA[wr * 32 + t * 16 + m16][0];
        af[t]  = mk_frag(*(const v8bf*)(pa + hi8), *(const v8bf*)(pa + 16 + hi8));
        const bf16_t* pb = &lB[wc * 32 + t * 16 + m16][0];
        bfr[t] = mk_frag(*(const v8bf*)(pb + hi8), *(const v8bf*)(pb + 16 + hi8));
      }
#pragma unroll
      for (int i = 0; i < 2; ++i)
#pragma unroll
        for (int j = 0; j < 2; ++j)
          acc[i][j] = __builtin_amdgcn_wmma_f32_16x16x32_bf16(
              false, af[i], false, bfr[j], (short)0, acc[i][j], false, false);
      __syncthreads();
    }
  } else {
    // ---------------- generic guarded path (ragged K / N) ----------------
    for (int k0 = 0; k0 < Kdim; k0 += GEMM_TK) {
      for (int i = tid; i < GEMM_TM * GEMM_TK; i += 256) {
        int r = i >> 5, c = i & 31;
        int gr = bM + r, gc = k0 + c;
        float v = 0.f;
        if (gr < M && gc < Kdim) {
          if constexpr (ABF) v = (float)((const bf16_t*)Aptr)[(size_t)gr * lda + gc];
          else               v = ((const float*)Aptr)[(size_t)gr * lda + gc];
        }
        lA[r][c] = (bf16_t)v;
      }
      for (int i = tid; i < GEMM_TN * GEMM_TK; i += 256) {
        int n, c;
        if constexpr (TRB) { n = i >> 5; c = i & 31; }
        else               { n = i & 63; c = i >> 6; }
        int gk = k0 + c, gn = bN + n;
        float v = 0.f;
        if (gk < Kdim && gn < N) {
          size_t idx = TRB ? ((size_t)gn * ldb + gk) : ((size_t)gk * ldb + gn);
          if constexpr (BBF) v = (float)((const bf16_t*)Bptr)[idx];
          else               v = ((const float*)Bptr)[idx];
        }
        lB[n][c] = (bf16_t)v;
      }
      __syncthreads();

      v16bf af[2], bfr[2];
#pragma unroll
      for (int t = 0; t < 2; ++t) {
        const bf16_t* pa = &lA[wr * 32 + t * 16 + m16][0];
        af[t]  = mk_frag(*(const v8bf*)(pa + hi8), *(const v8bf*)(pa + 16 + hi8));
        const bf16_t* pb = &lB[wc * 32 + t * 16 + m16][0];
        bfr[t] = mk_frag(*(const v8bf*)(pb + hi8), *(const v8bf*)(pb + 16 + hi8));
      }
#pragma unroll
      for (int i = 0; i < 2; ++i)
#pragma unroll
        for (int j = 0; j < 2; ++j)
          acc[i][j] = __builtin_amdgcn_wmma_f32_16x16x32_bf16(
              false, af[i], false, bfr[j], (short)0, acc[i][j], false, false);
      __syncthreads();
    }
  }

  // ---- epilogue (guard-free when AL: M%128==0 && N%64==0) ----
#pragma unroll
  for (int i = 0; i < 2; ++i)
#pragma unroll
    for (int j = 0; j < 2; ++j)
#pragma unroll
      for (int e = 0; e < 8; ++e) {
        int row = bM + wr * 32 + i * 16 + e + hi8;
        int col = bN + wc * 32 + j * 16 + m16;
        if (AL || (row < M && col < N)) {
          float v = acc[i][j][e] * alpha;
          if (bias) v += bias[col];
          if (epi == 1) {
            v = fmaxf(v, 0.f);
          } else if (epi == 2) {
            v = fmaxf(v, 0.f) + residual[(size_t)row * ldres + col];
            v = fmaxf(v, 0.f);
          }
          if (c_bf16) ((bf16_t*)Cptr)[(size_t)row * ldc + col] = (bf16_t)v;
          else        ((float*)Cptr)[(size_t)row * ldc + col] = v;
        }
      }
}

// ---------------------------------------------------------------------------
// Row softmax over Ncols (f32 in -> bf16 probs out).  One block per row.
// ---------------------------------------------------------------------------
__global__ __launch_bounds__(256) void softmax_rows_kernel(
    const float* __restrict__ S, bf16_t* __restrict__ P, int Ncols)
{
  __shared__ float red[256];
  const int row = blockIdx.x;
  const float* srow = S + (size_t)row * Ncols;
  float mx = -1e30f;
  for (int c = threadIdx.x; c < Ncols; c += 256) mx = fmaxf(mx, srow[c]);
  red[threadIdx.x] = mx; __syncthreads();
  for (int s = 128; s > 0; s >>= 1) {
    if (threadIdx.x < s) red[threadIdx.x] = fmaxf(red[threadIdx.x], red[threadIdx.x + s]);
    __syncthreads();
  }
  mx = red[0]; __syncthreads();
  float sum = 0.f;
  for (int c = threadIdx.x; c < Ncols; c += 256) sum += __expf(srow[c] - mx);
  red[threadIdx.x] = sum; __syncthreads();
  for (int s = 128; s > 0; s >>= 1) {
    if (threadIdx.x < s) red[threadIdx.x] += red[threadIdx.x + s];
    __syncthreads();
  }
  const float inv = 1.f / red[0];
  bf16_t* prow = P + (size_t)row * Ncols;
  for (int c = threadIdx.x; c < Ncols; c += 256)
    prow[c] = (bf16_t)(__expf(srow[c] - mx) * inv);
}

// ---------------------------------------------------------------------------
// Local group attention (16 points, d=512).  One block (4 waves) per group.
// Q/K staged row-padded (1040B stride -> 16 distinct banks for the 16 lanes);
// V staged transposed so the PV B-fragment is two contiguous b128 LDS loads.
// ---------------------------------------------------------------------------
__global__ __launch_bounds__(128) void local_attn_kernel(
    const bf16_t* __restrict__ Q, const bf16_t* __restrict__ Kmat,
    const bf16_t* __restrict__ V, bf16_t* __restrict__ O, float scale)
{
  __shared__ bf16_t sQ[16][520];      // padded rows
  __shared__ bf16_t sK[16][520];
  __shared__ bf16_t sVt[512][16];     // transposed V: sVt[n][k]
  __shared__ float  sS[4][16][16];
  __shared__ bf16_t sP[16][32];

  const int g = blockIdx.x;
  const size_t base = (size_t)g * 16 * 512;
  const int tid = threadIdx.x, wv = tid >> 5, lane = tid & 31;
  const int m16 = lane & 15, hi8 = (lane >> 4) * 8, kB = (lane >> 4) * 16;

  {
    const bf16_t* gq = Q + base;
    const bf16_t* gk = Kmat + base;
    const bf16_t* gv = V + base;
    for (int i = tid; i < 1024; i += 128) {          // 16 x 512 in 8-wide chunks
      int r = i >> 6, cv = (i & 63) * 8;
      *(v8bf*)&sQ[r][cv] = *(const v8bf*)(gq + r * 512 + cv);
      *(v8bf*)&sK[r][cv] = *(const v8bf*)(gk + r * 512 + cv);
    }
    for (int i = tid; i < 8192; i += 128) {          // transpose V while staging
      int r = i >> 9, c = i & 511;                    // coalesced along c
      sVt[c][r] = gv[r * 512 + c];
    }
  }
  __syncthreads();

  // S = Q K^T (partial over this wave's 128-deep K chunk)
  v8f s = zero8();
  for (int kk = wv * 128; kk < wv * 128 + 128; kk += 32) {
    v16bf a = mk_frag(*(const v8bf*)&sQ[m16][kk + hi8],
                      *(const v8bf*)&sQ[m16][kk + 16 + hi8]);
    v16bf b = mk_frag(*(const v8bf*)&sK[m16][kk + hi8],
                      *(const v8bf*)&sK[m16][kk + 16 + hi8]);
    s = __builtin_amdgcn_wmma_f32_16x16x32_bf16(false, a, false, b, (short)0, s, false, false);
  }
#pragma unroll
  for (int e = 0; e < 8; ++e) sS[wv][e + hi8][m16] = s[e];
  __syncthreads();
  {                                    // reduce 4 partials: 2 entries per thread
    int r = tid >> 4, c = tid & 15;    // rows 0..7
    sS[0][r][c] = (sS[0][r][c] + sS[1][r][c] + sS[2][r][c] + sS[3][r][c]) * scale;
    int r2 = r + 8;                    // rows 8..15
    sS[0][r2][c] = (sS[0][r2][c] + sS[1][r2][c] + sS[2][r2][c] + sS[3][r2][c]) * scale;
  }
  __syncthreads();
  if (tid < 16) {                      // softmax, one thread per row (16x16)
    int r = tid;
    float mx = -1e30f;
    for (int c = 0; c < 16; ++c) mx = fmaxf(mx, sS[0][r][c]);
    float ex[16], sum = 0.f;
    for (int c = 0; c < 16; ++c) { ex[c] = __expf(sS[0][r][c] - mx); sum += ex[c]; }
    float inv = 1.f / sum;
    for (int c = 0; c < 16; ++c)  sP[r][c] = (bf16_t)(ex[c] * inv);
    for (int c = 16; c < 32; ++c) sP[r][c] = (bf16_t)0.f;   // zero-pad K to 32
  }
  __syncthreads();

  // O = P V : this wave covers columns [wv*128, wv*128+128)
  v16bf a = mk_frag(*(const v8bf*)&sP[m16][kB],
                    *(const v8bf*)&sP[m16][kB + 8]);
  for (int nt = wv * 128; nt < wv * 128 + 128; nt += 16) {
    v16bf b;
    if (lane < 16) {                   // K rows 0..15 live on lanes 0..15
      b = mk_frag(*(const v8bf*)&sVt[nt + m16][0],
                  *(const v8bf*)&sVt[nt + m16][8]);
    } else {                           // K rows 16..31 are the zero pad
#pragma unroll
      for (int e = 0; e < 16; ++e) b[e] = (bf16_t)0.f;
    }
    v8f o = __builtin_amdgcn_wmma_f32_16x16x32_bf16(false, a, false, b, (short)0,
                                                    zero8(), false, false);
#pragma unroll
    for (int e = 0; e < 8; ++e)
      O[base + (size_t)(e + hi8) * 512 + nt + m16] = (bf16_t)o[e];
  }
}

// ---------------------------------------------------------------------------
// kNN top-16 (squared distance), one thread per query, LDS-tiled database.
// ---------------------------------------------------------------------------
#define KNN_K 16
__global__ __launch_bounds__(256) void knn_topk_kernel(
    const float* __restrict__ query, int qstride,
    const float* __restrict__ data,  int dstride,
    int Mq, int Nd, int* __restrict__ idx_out)
{
  __shared__ float tx[256], ty[256], tz[256];
  const int b = blockIdx.y;
  const int q = blockIdx.x * 256 + threadIdx.x;
  const bool active = q < Mq;
  float qx = 0.f, qy = 0.f, qz = 0.f;
  if (active) {
    const float* qp = query + ((size_t)b * Mq + q) * qstride;
    qx = qp[0]; qy = qp[1]; qz = qp[2];
  }
  float bestd[KNN_K]; int besti[KNN_K];
#pragma unroll
  for (int i = 0; i < KNN_K; ++i) { bestd[i] = 1e30f; besti[i] = 0; }

  for (int j0 = 0; j0 < Nd; j0 += 256) {
    int j = j0 + threadIdx.x;
    if (j < Nd) {
      const float* dp = data + ((size_t)b * Nd + j) * dstride;
      tx[threadIdx.x] = dp[0]; ty[threadIdx.x] = dp[1]; tz[threadIdx.x] = dp[2];
    }
    __syncthreads();
    int lim = Nd - j0; if (lim > 256) lim = 256;
    if (active) {
      for (int jj = 0; jj < lim; ++jj) {
        float dx = qx - tx[jj], dy = qy - ty[jj], dz = qz - tz[jj];
        float d = dx * dx + dy * dy + dz * dz;
        if (d < bestd[KNN_K - 1]) {
          int p = KNN_K - 1;
          while (p > 0 && bestd[p - 1] > d) {
            bestd[p] = bestd[p - 1]; besti[p] = besti[p - 1]; --p;
          }
          bestd[p] = d; besti[p] = j0 + jj;
        }
      }
    }
    __syncthreads();
  }
  if (active) {
    int* op = idx_out + ((size_t)b * Mq + q) * KNN_K;
#pragma unroll
    for (int i = 0; i < KNN_K; ++i) op[i] = besti[i];
  }
}

// ---------------------------------------------------------------------------
// Farthest point sampling: one block (1024 thr) per batch, serial argmax loop
// (matches the reference's scan dependence).  Records `far` BEFORE updating.
// ---------------------------------------------------------------------------
__global__ __launch_bounds__(1024) void fps_kernel(
    const float* __restrict__ xyz, int stride, int Npts, int npoint,
    int* __restrict__ idx_out)
{
  __shared__ float rd[1024];
  __shared__ int   ri[1024];
  __shared__ float cent[3];
  const int b = blockIdx.x;
  const float* base = xyz + (size_t)b * Npts * stride;
  float mind[4];
#pragma unroll
  for (int p = 0; p < 4; ++p) mind[p] = 1e10f;
  int far = 0;

  for (int it = 0; it < npoint; ++it) {
    if (threadIdx.x == 0) {
      idx_out[(size_t)b * npoint + it] = far;
      const float* fp = base + (size_t)far * stride;
      cent[0] = fp[0]; cent[1] = fp[1]; cent[2] = fp[2];
    }
    __syncthreads();
    const float cx = cent[0], cy = cent[1], cz = cent[2];
    float bm = -1.f; int bi = 0;
#pragma unroll
    for (int p = 0; p < 4; ++p) {
      int n = threadIdx.x + p * 1024;
      if (n < Npts) {
        const float* pp = base + (size_t)n * stride;
        float dx = pp[0] - cx, dy = pp[1] - cy, dz = pp[2] - cz;
        float d = dx * dx + dy * dy + dz * dz;
        if (d < mind[p]) mind[p] = d;
        if (mind[p] > bm) { bm = mind[p]; bi = n; }
      }
    }
    rd[threadIdx.x] = bm; ri[threadIdx.x] = bi;
    __syncthreads();
    for (int s = 512; s > 0; s >>= 1) {
      if (threadIdx.x < s && rd[threadIdx.x + s] > rd[threadIdx.x]) {
        rd[threadIdx.x] = rd[threadIdx.x + s]; ri[threadIdx.x] = ri[threadIdx.x + s];
      }
      __syncthreads();
    }
    far = ri[0];
    __syncthreads();
  }
}

// ---------------------------------------------------------------------------
// Small helpers
// ---------------------------------------------------------------------------
__global__ void extract_xyz_kernel(const float* __restrict__ x, int total,
                                   float* __restrict__ xyz)
{
  int t = blockIdx.x * 256 + threadIdx.x;
  if (t < total) {
    xyz[t * 3 + 0] = x[t * 6 + 0];
    xyz[t * 3 + 1] = x[t * 6 + 1];
    xyz[t * 3 + 2] = x[t * 6 + 2];
  }
}

__global__ void gather_xyz_kernel(const float* __restrict__ src, int Nsrc,
                                  const int* __restrict__ idx, int M,
                                  float* __restrict__ dst)
{
  int i = blockIdx.x * 256 + threadIdx.x;
  int b = blockIdx.y;
  if (i < M) {
    int s = idx[(size_t)b * M + i];
    const float* sp = src + ((size_t)b * Nsrc + s) * 3;
    float* dp = dst + ((size_t)b * M + i) * 3;
    dp[0] = sp[0]; dp[1] = sp[1]; dp[2] = sp[2];
  }
}

// g[(b,i,k), :] = { nbr_xyz - center_xyz (3), points[nbr] (C) }
__global__ void sa_group_kernel(const float* __restrict__ xyz, int Nd,
                                const float* __restrict__ nxyz,
                                const float* __restrict__ pts, int C,
                                const int* __restrict__ knn, int M, int Bc,
                                float* __restrict__ g)
{
  int t = blockIdx.x * 256 + threadIdx.x;
  int total = Bc * M * 16;
  if (t >= total) return;
  int k = t & 15;
  int i = (t >> 4) % M;
  int b = t / (M * 16);
  int nb = knn[((size_t)b * M + i) * 16 + k];
  const float* nx = nxyz + ((size_t)b * M + i) * 3;
  const float* dx = xyz + ((size_t)b * Nd + nb) * 3;
  const int din = 3 + C;
  float* gr = g + (size_t)t * din;
  gr[0] = dx[0] - nx[0]; gr[1] = dx[1] - nx[1]; gr[2] = dx[2] - nx[2];
  const float* pr = pts + ((size_t)b * Nd + nb) * C;
  for (int c = 0; c < C; ++c) gr[3 + c] = pr[c];
}

// Fused: gather neighbors, fc_delta (9->32)+relu, max over K, linear1+relu.
__global__ __launch_bounds__(256) void fc_delta_max_kernel(
    const float* __restrict__ x, const int* __restrict__ knn,
    const float* __restrict__ wd, const float* __restrict__ bd,
    const float* __restrict__ w1, const float* __restrict__ b1,
    int Npts, float* __restrict__ h)
{
  __shared__ float sWd[9 * 32], sBd[32], sW1[32 * 32], sB1[32];
  for (int i = threadIdx.x; i < 288; i += 256) sWd[i] = wd[i];
  for (int i = threadIdx.x; i < 1024; i += 256) sW1[i] = w1[i];
  if (threadIdx.x < 32) { sBd[threadIdx.x] = bd[threadIdx.x]; sB1[threadIdx.x] = b1[threadIdx.x]; }
  __syncthreads();

  const int b = blockIdx.y;
  const int n = blockIdx.x * 256 + threadIdx.x;
  if (n >= Npts) return;
  const float* xr = x + ((size_t)b * Npts + n) * 6;
  const float px = xr[0], py = xr[1], pz = xr[2];
  float hmax[32];
#pragma unroll
  for (int c = 0; c < 32; ++c) hmax[c] = -1e30f;
  const int* kr = knn + ((size_t)b * Npts + n) * 16;
  for (int k = 0; k < 16; ++k) {
    const float* nb = x + ((size_t)b * Npts + kr[k]) * 6;
    float f[9] = { px - nb[0], py - nb[1], pz - nb[2],
                   nb[0], nb[1], nb[2], nb[3], nb[4], nb[5] };
#pragma unroll
    for (int c = 0; c < 32; ++c) {
      float a = sBd[c];
#pragma unroll
      for (int r = 0; r < 9; ++r) a += f[r] * sWd[r * 32 + c];
      a = fmaxf(a, 0.f);
      if (a > hmax[c]) hmax[c] = a;
    }
  }
  float* hr = h + ((size_t)b * Npts + n) * 32;
#pragma unroll
  for (int c = 0; c < 32; ++c) {
    float a = sB1[c];
#pragma unroll
    for (int r = 0; r < 32; ++r) a += hmax[r] * sW1[r * 32 + c];
    hr[c] = fmaxf(a, 0.f);
  }
}

// conv0+bn+relu -> conv1+bn+relu -> max over K=16.  One wave per point,
// channels split across lanes, conv0 activations shared via LDS.
__global__ __launch_bounds__(128) void convs_max_kernel(
    const float* __restrict__ g,
    const float* __restrict__ w0, const float* __restrict__ b0,
    const float* __restrict__ gm0, const float* __restrict__ bt0,
    const float* __restrict__ w1, const float* __restrict__ b1,
    const float* __restrict__ gm1, const float* __restrict__ bt1,
    int din, int c0, int c1, int total, float* __restrict__ out)
{
  __shared__ float sT0[4][128];
  const int wv = threadIdx.x >> 5, lane = threadIdx.x & 31;
  int pt = blockIdx.x * 4 + wv;
  const bool valid = pt < total;
  if (!valid) pt = total - 1;                 // keep uniform control flow
  const float bnscale = rsqrtf(1.f + 1e-5f);  // matches _bn_eval
  float best[4];
#pragma unroll
  for (int i = 0; i < 4; ++i) best[i] = -1e30f;

  for (int k = 0; k < 16; ++k) {
    const float* gr = g + ((size_t)pt * 16 + k) * din;
#pragma unroll
    for (int ci = 0; ci < 4; ++ci) {
      int c = lane + ci * 32;
      if (c < c0) {
        float a = b0[c];
        for (int r = 0; r < din; ++r) a += gr[r] * w0[(size_t)r * c0 + c];
        a = a * (gm0[c] * bnscale) + bt0[c];
        sT0[wv][c] = fmaxf(a, 0.f);
      }
    }
    __syncthreads();
#pragma unroll
    for (int ci = 0; ci < 4; ++ci) {
      int c = lane + ci * 32;
      if (c < c1) {
        float a = b1[c];
        for (int r = 0; r < c0; ++r) a += sT0[wv][r] * w1[(size_t)r * c1 + c];
        a = a * (gm1[c] * bnscale) + bt1[c];
        a = fmaxf(a, 0.f);
        if (a > best[ci]) best[ci] = a;
      }
    }
    __syncthreads();
  }
  if (valid) {
#pragma unroll
    for (int ci = 0; ci < 4; ++ci) {
      int c = lane + ci * 32;
      if (c < c1) out[(size_t)pt * c1 + c] = best[ci];
    }
  }
}

// ---------------------------------------------------------------------------
// Orchestration
// ---------------------------------------------------------------------------
extern "C" void kernel_launch(void* const* d_in, const int* in_sizes, int n_in,
                              void* d_out, int out_size, void* d_ws, size_t ws_size,
                              hipStream_t stream)
{
  (void)in_sizes; (void)n_in; (void)out_size; (void)ws_size;

  // -- unpack inputs: setup_inputs() dict order (insertion-order pytree) --
  int p = 0;
  const float* x        = (const float*)d_in[p++];  // (4,4096,6)
  const float* fcd_w    = (const float*)d_in[p++];  // 9x32
  const float* fcd_b    = (const float*)d_in[p++];
  const float* lin1_w   = (const float*)d_in[p++];  // 32x32
  const float* lin1_b   = (const float*)d_in[p++];
  const float* tr1_fc1w = (const float*)d_in[p++];  // 32x512
  const float* tr1_fc1b = (const float*)d_in[p++];
  const float* tr1_wq   = (const float*)d_in[p++];  // 512x512
  const float* tr1_wk   = (const float*)d_in[p++];
  const float* tr1_wv   = (const float*)d_in[p++];
  const float* tr1_fc2w = (const float*)d_in[p++];  // 512x32
  const float* tr1_fc2b = (const float*)d_in[p++];
  const float* s0_fc1w  = (const float*)d_in[p++];  // 35x512
  const float* s0_fc1b  = (const float*)d_in[p++];
  const float* s0_wq    = (const float*)d_in[p++];
  const float* s0_wk    = (const float*)d_in[p++];
  const float* s0_wv    = (const float*)d_in[p++];
  const float* s0_fc2w  = (const float*)d_in[p++];  // 512x35
  const float* s0_fc2b  = (const float*)d_in[p++];
  const float* s0_c0w   = (const float*)d_in[p++];  // 35x64
  const float* s0_c0b   = (const float*)d_in[p++];
  const float* s0_c0g   = (const float*)d_in[p++];
  const float* s0_c0be  = (const float*)d_in[p++];
  const float* s0_c1w   = (const float*)d_in[p++];  // 64x64
  const float* s0_c1b   = (const float*)d_in[p++];
  const float* s0_c1g   = (const float*)d_in[p++];
  const float* s0_c1be  = (const float*)d_in[p++];
  const float* s1_fc1w  = (const float*)d_in[p++];  // 67x512
  const float* s1_fc1b  = (const float*)d_in[p++];
  const float* s1_wq    = (const float*)d_in[p++];
  const float* s1_wk    = (const float*)d_in[p++];
  const float* s1_wv    = (const float*)d_in[p++];
  const float* s1_fc2w  = (const float*)d_in[p++];  // 512x67
  const float* s1_fc2b  = (const float*)d_in[p++];
  const float* s1_c0w   = (const float*)d_in[p++];  // 67x128
  const float* s1_c0b   = (const float*)d_in[p++];
  const float* s1_c0g   = (const float*)d_in[p++];
  const float* s1_c0be  = (const float*)d_in[p++];
  const float* s1_c1w   = (const float*)d_in[p++];  // 128x128
  const float* s1_c1b   = (const float*)d_in[p++];
  const float* s1_c1g   = (const float*)d_in[p++];
  const float* s1_c1be  = (const float*)d_in[p++];

  const int Bb = 4, Npts = 4096;
  const float scale = 0.044194173824159216f;  // 1/sqrt(512)

  // -- workspace layout (bump allocator, ~460 MB) --
  char* ws = (char*)d_ws;
  size_t off = 0;
  auto alloc = [&](size_t bytes) -> char* {
    char* r = ws + off;
    off += (bytes + 255) & ~(size_t)255;
    return r;
  };
  float*  xyz  = (float*) alloc((size_t)Bb * Npts * 3 * 4);
  float*  hbuf = (float*) alloc((size_t)Bb * Npts * 32 * 4);
  float*  pts1 = (float*) alloc((size_t)Bb * Npts * 32 * 4);
  bf16_t* Xb   = (bf16_t*)alloc((size_t)65536 * 512 * 2);
  bf16_t* Qb   = (bf16_t*)alloc((size_t)65536 * 512 * 2);
  bf16_t* Kb   = (bf16_t*)alloc((size_t)65536 * 512 * 2);
  bf16_t* Vb   = (bf16_t*)alloc((size_t)65536 * 512 * 2);
  bf16_t* Ob   = (bf16_t*)alloc((size_t)65536 * 512 * 2);
  float*  Sb   = (float*) alloc((size_t)4096 * 4096 * 4);
  bf16_t* Pb   = (bf16_t*)alloc((size_t)4096 * 4096 * 2);
  float*  g0   = (float*) alloc((size_t)65536 * 67 * 4);
  float*  g1   = (float*) alloc((size_t)65536 * 67 * 4);
  int*    fpsi = (int*)   alloc((size_t)Bb * 1024 * 4);
  float*  nx0  = (float*) alloc((size_t)Bb * 1024 * 3 * 4);
  float*  nx1  = (float*) alloc((size_t)Bb * 256 * 3 * 4);
  int*    knni = (int*)   alloc((size_t)Bb * Npts * 16 * 4);
  float*  pts0 = (float*) alloc((size_t)Bb * 1024 * 64 * 4);

  // Only the (dtype, layout) combinations this model actually launches are
  // instantiated as fast kernels; anything else falls back to a guarded
  // generic instantiation so every runtime combination still works.
  auto gemm = [&](const void* A, int a_bf16, int lda,
                  const void* Bm, int b_bf16, int ldb, int transB,
                  const float* bias, const float* res, int ldres,
                  void* C, int c_bf16, int ldc,
                  int M, int N, int Kd, float alpha, int epi) {
    dim3 grid((N + GEMM_TN - 1) / GEMM_TN, (M + GEMM_TM - 1) / GEMM_TM);
    bool aligned = (M % GEMM_TM == 0) && (N % GEMM_TN == 0) && (Kd % GEMM_TK == 0)
                   && (a_bf16 ? (lda % 8 == 0) : (lda % 4 == 0))
                   && (!(transB && b_bf16) || (ldb % 8 == 0));
#define GEMM_GO(AB, BB, TB, ALG)                                                \
    gemm_wmma_t<AB, BB, TB, ALG><<<grid, 256, 0, stream>>>(                     \
        A, lda, Bm, ldb, bias, res, ldres, C, c_bf16, ldc, M, N, Kd, alpha, epi)
    if (aligned && !a_bf16 && !b_bf16 && !transB)      GEMM_GO(false, false, false, true);
    else if (aligned && a_bf16 && !b_bf16 && !transB)  GEMM_GO(true,  false, false, true);
    else if (aligned && a_bf16 &&  b_bf16 &&  transB)  GEMM_GO(true,  true,  true,  true);
    else if (aligned && a_bf16 &&  b_bf16 && !transB)  GEMM_GO(true,  true,  false, true);
    else if (!a_bf16 && !b_bf16 && !transB)            GEMM_GO(false, false, false, false);
    else if (a_bf16 && !b_bf16 && !transB)             GEMM_GO(true,  false, false, false);
    else if (a_bf16 &&  b_bf16 &&  transB)             GEMM_GO(true,  true,  true,  false);
    else                                               GEMM_GO(true,  true,  false, false);
#undef GEMM_GO
  };

  // ---- stage 0: xyz extraction, self-kNN, fc_delta+max+linear1 ----
  extract_xyz_kernel<<<(Bb * Npts + 255) / 256, 256, 0, stream>>>(x, Bb * Npts, xyz);
  knn_topk_kernel<<<dim3(Npts / 256, Bb), 256, 0, stream>>>(xyz, 3, xyz, 3, Npts, Npts, knni);
  fc_delta_max_kernel<<<dim3(Npts / 256, Bb), 256, 0, stream>>>(
      x, knni, fcd_w, fcd_b, lin1_w, lin1_b, Npts, hbuf);

  // ---- stage 1: global transformer tr1 (N=4096, d=512) ----
  const int M1 = Bb * Npts;  // 16384
  gemm(hbuf, 0, 32,  tr1_fc1w, 0, 512, 0, tr1_fc1b, nullptr, 0, Xb, 1, 512, M1, 512, 32, 1.f, 0);
  gemm(Xb, 1, 512, tr1_wq, 0, 512, 0, nullptr, nullptr, 0, Qb, 1, 512, M1, 512, 512, 1.f, 0);
  gemm(Xb, 1, 512, tr1_wk, 0, 512, 0, nullptr, nullptr, 0, Kb, 1, 512, M1, 512, 512, 1.f, 0);
  gemm(Xb, 1, 512, tr1_wv, 0, 512, 0, nullptr, nullptr, 0, Vb, 1, 512, M1, 512, 512, 1.f, 0);
  for (int b = 0; b < Bb; ++b) {  // S, softmax, PV per batch (S stays L2-resident)
    const bf16_t* Qp = Qb + (size_t)b * Npts * 512;
    const bf16_t* Kp = Kb + (size_t)b * Npts * 512;
    const bf16_t* Vp = Vb + (size_t)b * Npts * 512;
    bf16_t*       Op = Ob + (size_t)b * Npts * 512;
    gemm(Qp, 1, 512, Kp, 1, 512, 1, nullptr, nullptr, 0, Sb, 0, Npts, Npts, Npts, 512, scale, 0);
    softmax_rows_kernel<<<Npts, 256, 0, stream>>>(Sb, Pb, Npts);
    gemm(Pb, 1, Npts, Vp, 1, 512, 0, nullptr, nullptr, 0, Op, 1, 512, Npts, 512, Npts, 1.f, 0);
  }
  // pts1 = relu(h + relu(O @ fc2 + b))
  gemm(Ob, 1, 512, tr1_fc2w, 0, 32, 0, tr1_fc2b, hbuf, 32, pts1, 0, 32, M1, 32, 512, 1.f, 2);

  // ---- stage 2: sa0 (4096 -> 1024, d_in=35, convs 64/64) ----
  fps_kernel<<<Bb, 1024, 0, stream>>>(xyz, 3, Npts, 1024, fpsi);
  gather_xyz_kernel<<<dim3(4, Bb), 256, 0, stream>>>(xyz, Npts, fpsi, 1024, nx0);
  knn_topk_kernel<<<dim3(4, Bb), 256, 0, stream>>>(nx0, 3, xyz, 3, 1024, Npts, knni);
  sa_group_kernel<<<(Bb * 1024 * 16 + 255) / 256, 256, 0, stream>>>(
      xyz, Npts, nx0, pts1, 32, knni, 1024, Bb, g0);
  const int M0 = Bb * 1024 * 16;  // 65536
  gemm(g0, 0, 35,  s0_fc1w, 0, 512, 0, s0_fc1b, nullptr, 0, Xb, 1, 512, M0, 512, 35, 1.f, 0);
  gemm(Xb, 1, 512, s0_wq,   0, 512, 0, nullptr, nullptr, 0, Qb, 1, 512, M0, 512, 512, 1.f, 0);
  gemm(Xb, 1, 512, s0_wk,   0, 512, 0, nullptr, nullptr, 0, Kb, 1, 512, M0, 512, 512, 1.f, 0);
  gemm(Xb, 1, 512, s0_wv,   0, 512, 0, nullptr, nullptr, 0, Vb, 1, 512, M0, 512, 512, 1.f, 0);
  local_attn_kernel<<<Bb * 1024, 128, 0, stream>>>(Qb, Kb, Vb, Ob, scale);
  gemm(Ob, 1, 512, s0_fc2w, 0, 35, 0, s0_fc2b, nullptr, 0, g1, 0, 35, M0, 35, 512, 1.f, 1);
  convs_max_kernel<<<(Bb * 1024) / 4, 128, 0, stream>>>(
      g1, s0_c0w, s0_c0b, s0_c0g, s0_c0be, s0_c1w, s0_c1b, s0_c1g, s0_c1be,
      35, 64, 64, Bb * 1024, pts0);

  // ---- stage 3: sa1 (1024 -> 256, d_in=67, convs 128/128) ----
  fps_kernel<<<Bb, 1024, 0, stream>>>(nx0, 3, 1024, 256, fpsi);
  gather_xyz_kernel<<<dim3(1, Bb), 256, 0, stream>>>(nx0, 1024, fpsi, 256, nx1);
  knn_topk_kernel<<<dim3(1, Bb), 256, 0, stream>>>(nx1, 3, nx0, 3, 256, 1024, knni);
  sa_group_kernel<<<(Bb * 256 * 16 + 255) / 256, 256, 0, stream>>>(
      nx0, 1024, nx1, pts0, 64, knni, 256, Bb, g0);
  const int M2 = Bb * 256 * 16;  // 16384
  gemm(g0, 0, 67,  s1_fc1w, 0, 512, 0, s1_fc1b, nullptr, 0, Xb, 1, 512, M2, 512, 67, 1.f, 0);
  gemm(Xb, 1, 512, s1_wq,   0, 512, 0, nullptr, nullptr, 0, Qb, 1, 512, M2, 512, 512, 1.f, 0);
  gemm(Xb, 1, 512, s1_wk,   0, 512, 0, nullptr, nullptr, 0, Kb, 1, 512, M2, 512, 512, 1.f, 0);
  gemm(Xb, 1, 512, s1_wv,   0, 512, 0, nullptr, nullptr, 0, Vb, 1, 512, M2, 512, 512, 1.f, 0);
  local_attn_kernel<<<Bb * 256, 128, 0, stream>>>(Qb, Kb, Vb, Ob, scale);
  gemm(Ob, 1, 512, s1_fc2w, 0, 67, 0, s1_fc2b, nullptr, 0, g1, 0, 67, M2, 67, 512, 1.f, 1);
  convs_max_kernel<<<(Bb * 256) / 4, 128, 0, stream>>>(
      g1, s1_c0w, s1_c0b, s1_c0g, s1_c0be, s1_c1w, s1_c1b, s1_c1g, s1_c1be,
      67, 128, 128, Bb * 256, (float*)d_out);
}
